// MultiHeadSelfAttention_49478023250621
// MI455X (gfx1250) — matrix operand order repaired and checked by reading, not compiled, not run on hardware
//
#include <hip/hip_runtime.h>
#include <math.h>

#define D_MODEL 1024
#define NHEAD   16
#define DK      64
#define TSEQ    2048
#define BATCH   4
#define NTOK    (BATCH * TSEQ)   // 8192

// 0.125 * log2(e): QK^T scores land directly in the exp2 domain
#define QSCALE 0.18033688011118324f

typedef __attribute__((ext_vector_type(16))) __bf16 v16bf;
typedef __attribute__((ext_vector_type(8)))  float  v8f;
typedef __attribute__((ext_vector_type(8)))  __bf16 bf16x8;

union FragU { v16bf v; bf16x8 h[2]; };

__device__ __forceinline__ __bf16 f2bf(float x) { return (__bf16)x; }  // native cvt

// A-fragment (16x32, 16-bit): lane<16: K0-7 (V0-3), K16-23 (V4-7); lane>=16: K8-15, K24-31.
__device__ __forceinline__ v16bf lds_fragA(const __bf16* base, int pitch, int lane) {
    const int half = lane >> 4, l = lane & 15;
    const __bf16* r = base + l * pitch + half * 8;
    FragU f;
    f.h[0] = *(const bf16x8*)(r);
    f.h[1] = *(const bf16x8*)(r + 16);
    return f.v;
}

// B-fragment (32x16, 16-bit): lane = N; lane<16: K0-7, K16-23; lane>=16: K8-15, K24-31.
__device__ __forceinline__ v16bf lds_fragB(const __bf16* base, int pitch, int lane) {
    const __bf16* r = base + (lane & 15) * pitch + (lane >> 4) * 8;
    FragU f;
    f.h[0] = *(const bf16x8*)(r);
    f.h[1] = *(const bf16x8*)(r + 16);
    return f.v;
}

__device__ __forceinline__ v16bf ones_frag() {     // B-frag of all 1.0 (bf16 0x3F80)
    union { unsigned short s[16]; v16bf v; } u;
#pragma unroll
    for (int i = 0; i < 16; ++i) u.s[i] = 0x3F80;
    return u.v;
}

// gfx1250 async copy: LDS[lds] = MEM[g] (16B per lane), tracked by ASYNCcnt
__device__ __forceinline__ void async_b128(const void* g, void* lds) {
    unsigned l32 = (unsigned)(size_t)lds;
    unsigned long long g64 = (unsigned long long)(size_t)g;
    asm volatile("global_load_async_to_lds_b128 %0, %1, off"
                 :: "v"(l32), "v"(g64) : "memory");
}
#define WAIT_ASYNC_8() asm volatile("s_wait_asynccnt 0x8" ::: "memory")
#define WAIT_ASYNC_0() asm volatile("s_wait_asynccnt 0x0" ::: "memory")

// ---------------- fp32 -> bf16 convert pass ----------------
__global__ __launch_bounds__(256)
void cvt_kernel(const float* __restrict__ in, __bf16* __restrict__ out, int n) {
    int i = (blockIdx.x * 256 + threadIdx.x) * 8;
    if (i >= n) return;
    const float4* s = (const float4*)(in + i);
    float4 a = s[0], b = s[1];
    bf16x8 o;
    o[0] = f2bf(a.x); o[1] = f2bf(a.y); o[2] = f2bf(a.z); o[3] = f2bf(a.w);
    o[4] = f2bf(b.x); o[5] = f2bf(b.y); o[6] = f2bf(b.z); o[7] = f2bf(b.w);
    *(bf16x8*)(out + i) = o;
}

// ---------------- GEMM: C[row,col] = sum_c A[row,c]*W[col,c] + bias[col] ----------------
// MODE 0: bf16 Q [b,h,t,dk] * QSCALE | 1: bf16 K [b,h,t,dk] | 2: bf16 V^T [b,h,dk,t] | 3: fp32 [row,col]
#define GP 72
#define NKB (D_MODEL / 64)

template<int MODE>
__global__ __launch_bounds__(256)
void mha_gemm_kernel(const __bf16* __restrict__ A, const __bf16* __restrict__ W,
                     const float* __restrict__ bias, void* __restrict__ Out)
{
    __shared__ __bf16 As[2][128 * GP];
    __shared__ __bf16 Ws[2][128 * GP];

    const int tid  = threadIdx.x;
    const int lane = tid & 31;
    const int wave = tid >> 5;
    const int waveM = wave & 3;
    const int waveN = wave >> 2;
    const int bm = blockIdx.y;
    const int bn = blockIdx.x;

    const int srow = tid >> 1;
    const int scol = (tid & 1) * 32;
    const __bf16* gA = A + (size_t)(bm * 128 + srow) * D_MODEL + scol;
    const __bf16* gW = W + (size_t)(bn * 128 + srow) * D_MODEL + scol;

    v8f acc[2][4] = {};

#pragma unroll
    for (int st = 0; st < 2; ++st)
#pragma unroll
        for (int j = 0; j < 4; ++j) {
            async_b128(gA + st * 64 + j * 8, &As[st][srow * GP + scol] + j * 8);
            async_b128(gW + st * 64 + j * 8, &Ws[st][srow * GP + scol] + j * 8);
        }

    for (int kb = 0; kb < NKB; ++kb) {
        if (kb + 1 < NKB) { WAIT_ASYNC_8(); } else { WAIT_ASYNC_0(); }
        __syncthreads();

        const int buf = kb & 1;
        const __bf16* At = &As[buf][(waveM * 32) * GP];
        const __bf16* Wt = &Ws[buf][(waveN * 64) * GP];
#pragma unroll
        for (int kk = 0; kk < 2; ++kk) {
            v16bf afr[2], bfr[4];
#pragma unroll
            for (int mt = 0; mt < 2; ++mt)
                afr[mt] = lds_fragA(At + (mt * 16) * GP + kk * 32, GP, lane);
#pragma unroll
            for (int nt = 0; nt < 4; ++nt)
                bfr[nt] = lds_fragB(Wt + (nt * 16) * GP + kk * 32, GP, lane);
#pragma unroll
            for (int mt = 0; mt < 2; ++mt)
#pragma unroll
                for (int nt = 0; nt < 4; ++nt)
                    acc[mt][nt] = __builtin_amdgcn_wmma_f32_16x16x32_bf16(
                        false, afr[mt], false, bfr[nt], (short)0, acc[mt][nt], false, false);
        }
        __syncthreads();

        if (kb + 2 < NKB) {
#pragma unroll
            for (int j = 0; j < 4; ++j) {
                async_b128(gA + (kb + 2) * 64 + j * 8, &As[buf][srow * GP + scol] + j * 8);
                async_b128(gW + (kb + 2) * 64 + j * 8, &Ws[buf][srow * GP + scol] + j * 8);
            }
        }
    }

    const int half = lane >> 4, ln = lane & 15;
#pragma unroll
    for (int nt = 0; nt < 4; ++nt) {
        const int col = bn * 128 + waveN * 64 + nt * 16 + ln;
        const float bc = bias[col];
        const int h = col >> 6, d = col & (DK - 1);
#pragma unroll
        for (int mt = 0; mt < 2; ++mt) {
            const int rowbase = bm * 128 + waveM * 32 + mt * 16 + half * 8;
            const int b = rowbase >> 11, t0 = rowbase & (TSEQ - 1);
            if (MODE == 2) {
                bf16x8 pk;
#pragma unroll
                for (int r = 0; r < 8; ++r) pk[r] = f2bf(acc[mt][nt][r] + bc);
                *(bf16x8*)((__bf16*)Out + (((size_t)(b * NHEAD + h)) * DK + d) * TSEQ + t0) = pk;
            } else {
#pragma unroll
                for (int r = 0; r < 8; ++r) {
                    const float val = acc[mt][nt][r] + bc;
                    if (MODE == 3)
                        ((float*)Out)[(size_t)(rowbase + r) * D_MODEL + col] = val;
                    else if (MODE == 0)
                        ((__bf16*)Out)[(((size_t)(b * NHEAD + h)) * TSEQ + t0 + r) * DK + d] = f2bf(val * QSCALE);
                    else
                        ((__bf16*)Out)[(((size_t)(b * NHEAD + h)) * TSEQ + t0 + r) * DK + d] = f2bf(val);
                }
            }
        }
    }
}

// ---------------- flash attention (128-key chunks) ----------------
#define KP  72                 // Ks pitch (dk direction)
#define VP  136                // Vs / Ps pitch (key direction)
#define KCH 128
#define NKC (TSEQ / KCH)       // 16

__global__ __launch_bounds__(256)
void mha_attn_kernel(const __bf16* __restrict__ Q, const __bf16* __restrict__ K,
                     const __bf16* __restrict__ VT, __bf16* __restrict__ Aout)
{
    __shared__ __bf16 Ks[2][KCH * KP];   // 128 keys x 64 dk
    __shared__ __bf16 Vs[2][64 * VP];    // 64 d x 128 keys
    __shared__ __bf16 Ps[8 * 16 * VP];   // per-wave 16 q x 128 keys

    const int tid  = threadIdx.x;
    const int lane = tid & 31;
    const int wave = tid >> 5;
    const int bh   = blockIdx.y;
    const int qbase = blockIdx.x * 128 + wave * 16;
    const int half = lane >> 4, ln = lane & 15;

    // Q A-fragments for dk 0..31 / 32..63
    v16bf qf0, qf1;
    {
        const __bf16* qrow = Q + ((size_t)bh * TSEQ + qbase + ln) * DK;
        FragU f0, f1;
        f0.h[0] = *(const bf16x8*)(qrow + half * 8);
        f0.h[1] = *(const bf16x8*)(qrow + 16 + half * 8);
        f1.h[0] = *(const bf16x8*)(qrow + 32 + half * 8);
        f1.h[1] = *(const bf16x8*)(qrow + 48 + half * 8);
        qf0 = f0.v; qf1 = f1.v;
    }
    const v16bf onesB = ones_frag();

    v8f o[4] = {};
    float mrow[8], lsum[8];
#pragma unroll
    for (int r = 0; r < 8; ++r) { mrow[r] = -1e30f; lsum[r] = 0.f; }

    __bf16* Pw = &Ps[wave * 16 * VP];

    // staging maps: K tile: row=tid>>1 (0..127), 32-elem chunk; V tile: row=tid>>2 (0..63), 32-elem chunk
    const int krow = tid >> 1, kcc = (tid & 1) * 32;
    const int vrow = tid >> 2, vcc = (tid & 3) * 32;
    const __bf16* gK = K  + ((size_t)bh * TSEQ + krow) * DK + kcc;
    const __bf16* gV = VT + ((size_t)bh * DK + vrow) * TSEQ + vcc;

#pragma unroll
    for (int st = 0; st < 2; ++st)
#pragma unroll
        for (int j = 0; j < 4; ++j) {
            async_b128(gK + (size_t)st * KCH * DK + j * 8, &Ks[st][krow * KP + kcc] + j * 8);
            async_b128(gV + st * KCH + j * 8,              &Vs[st][vrow * VP + vcc] + j * 8);
        }

    for (int kc = 0; kc < NKC; ++kc) {
        if (kc + 1 < NKC) { WAIT_ASYNC_8(); } else { WAIT_ASYNC_0(); }
        __syncthreads();
        const int buf = kc & 1;

        // S = Q K^T : 8 key tiles, 2 WMMA each (scores arrive pre-scaled into exp2 domain)
        v8f s[8];
#pragma unroll
        for (int nt = 0; nt < 8; ++nt) {
            v16bf k0 = lds_fragB(&Ks[buf][(nt * 16) * KP], KP, lane);
            v16bf k1 = lds_fragB(&Ks[buf][(nt * 16) * KP] + 32, KP, lane);
            v8f a = {};
            a = __builtin_amdgcn_wmma_f32_16x16x32_bf16(false, qf0, false, k0, (short)0, a, false, false);
            a = __builtin_amdgcn_wmma_f32_16x16x32_bf16(false, qf1, false, k1, (short)0, a, false, false);
            s[nt] = a;
        }

        // row max (16-lane halves hold one row each per VGPR index)
        float mnew[8], corr[8];
#pragma unroll
        for (int r = 0; r < 8; ++r) {
            float m = fmaxf(fmaxf(fmaxf(s[0][r], s[1][r]), fmaxf(s[2][r], s[3][r])),
                            fmaxf(fmaxf(s[4][r], s[5][r]), fmaxf(s[6][r], s[7][r])));
#pragma unroll
            for (int off = 1; off < 16; off <<= 1)
                m = fmaxf(m, __shfl_xor(m, off, 32));
            mnew[r] = fmaxf(mrow[r], m);
            corr[r] = exp2f(mrow[r] - mnew[r]);
            mrow[r] = mnew[r];
#pragma unroll
            for (int dt = 0; dt < 4; ++dt) o[dt][r] *= corr[r];
        }

        // P = exp2(S - m), stored to per-wave LDS tile (C layout -> row-major)
#pragma unroll
        for (int nt = 0; nt < 8; ++nt)
#pragma unroll
            for (int r = 0; r < 8; ++r)
                Pw[(r + half * 8) * VP + nt * 16 + ln] = f2bf(exp2f(s[nt][r] - mnew[r]));

        // reload P as A-fragments (DS ops are in-order within a wave)
        v16bf pf[4];
#pragma unroll
        for (int kk = 0; kk < 4; ++kk)
            pf[kk] = lds_fragA(Pw + kk * 32, VP, lane);

        // row sums via tensor core: rsum = P x ones
        v8f rsum = {};
#pragma unroll
        for (int kk = 0; kk < 4; ++kk)
            rsum = __builtin_amdgcn_wmma_f32_16x16x32_bf16(
                false, pf[kk], false, onesB, (short)0, rsum, false, false);
#pragma unroll
        for (int r = 0; r < 8; ++r)
            lsum[r] = lsum[r] * corr[r] + rsum[r];

        // O += P V
#pragma unroll
        for (int dt = 0; dt < 4; ++dt) {
#pragma unroll
            for (int kk = 0; kk < 4; ++kk) {
                v16bf vf = lds_fragB(&Vs[buf][(dt * 16) * VP] + kk * 32, VP, lane);
                o[dt] = __builtin_amdgcn_wmma_f32_16x16x32_bf16(
                    false, pf[kk], false, vf, (short)0, o[dt], false, false);
            }
        }
        __syncthreads();                         // everyone done with Ks/Vs[buf]

        if (kc + 2 < NKC) {
#pragma unroll
            for (int j = 0; j < 4; ++j) {
                async_b128(gK + (size_t)(kc + 2) * KCH * DK + j * 8, &Ks[buf][krow * KP + kcc] + j * 8);
                async_b128(gV + (kc + 2) * KCH + j * 8,              &Vs[buf][vrow * VP + vcc] + j * 8);
            }
        }
    }

    // normalize, write bf16 attn output [token, h*64+d]
    const int bb = bh >> 4, h = bh & 15;
#pragma unroll
    for (int r = 0; r < 8; ++r) {
        const float inv = 1.0f / lsum[r];
        const int t = qbase + r + half * 8;
#pragma unroll
        for (int dt = 0; dt < 4; ++dt)
            Aout[((size_t)(bb * TSEQ + t)) * D_MODEL + h * DK + dt * 16 + ln] =
                f2bf(o[dt][r] * inv);
    }
}

extern "C" void kernel_launch(void* const* d_in, const int* in_sizes, int n_in,
                              void* d_out, int out_size, void* d_ws, size_t ws_size,
                              hipStream_t stream) {
    (void)in_sizes; (void)n_in; (void)out_size; (void)ws_size;
    const float* x  = (const float*)d_in[0];
    const float* wq = (const float*)d_in[1];
    const float* bq = (const float*)d_in[2];
    const float* wk = (const float*)d_in[3];
    const float* bk = (const float*)d_in[4];
    const float* wv = (const float*)d_in[5];
    const float* bv = (const float*)d_in[6];
    const float* wo = (const float*)d_in[7];
    const float* bo = (const float*)d_in[8];

    const size_t xe = (size_t)NTOK * D_MODEL;
    const size_t we = (size_t)D_MODEL * D_MODEL;
    __bf16* xb  = (__bf16*)d_ws;
    __bf16* wqb = xb  + xe;
    __bf16* wkb = wqb + we;
    __bf16* wvb = wkb + we;
    __bf16* wob = wvb + we;
    __bf16* q   = wob + we;
    __bf16* k   = q   + xe;
    __bf16* vt  = k   + xe;
    __bf16* ao  = vt  + xe;

    cvt_kernel<<<dim3((int)(xe / 2048)), 256, 0, stream>>>(x,  xb,  (int)xe);
    cvt_kernel<<<dim3((int)(we / 2048)), 256, 0, stream>>>(wq, wqb, (int)we);
    cvt_kernel<<<dim3((int)(we / 2048)), 256, 0, stream>>>(wk, wkb, (int)we);
    cvt_kernel<<<dim3((int)(we / 2048)), 256, 0, stream>>>(wv, wvb, (int)we);
    cvt_kernel<<<dim3((int)(we / 2048)), 256, 0, stream>>>(wo, wob, (int)we);

    dim3 ggrid(D_MODEL / 128, NTOK / 128);           // (8, 64)
    mha_gemm_kernel<0><<<ggrid, 256, 0, stream>>>(xb, wqb, bq, q);
    mha_gemm_kernel<1><<<ggrid, 256, 0, stream>>>(xb, wkb, bk, k);
    mha_gemm_kernel<2><<<ggrid, 256, 0, stream>>>(xb, wvb, bv, vt);

    dim3 agrid(TSEQ / 128, BATCH * NHEAD);           // (16, 64)
    mha_attn_kernel<<<agrid, 256, 0, stream>>>(q, k, vt, ao);

    mha_gemm_kernel<3><<<ggrid, 256, 0, stream>>>(ao, wob, bo, d_out);
}
